// MANNBaseEncoder_49297634623749
// MI455X (gfx1250) — matrix operand order repaired and checked by reading, not compiled, not run on hardware
//
#include <hip/hip_runtime.h>

typedef __bf16 bf16_t;
typedef __attribute__((ext_vector_type(16))) __bf16 v16bf;
typedef __attribute__((ext_vector_type(8)))  float  v8f;

#define IDIM 512
#define CDIM 512
#define NMEM 1024
#define MW   64
#define TT   256
#define BSZ  128
#define KC   1088            // IDIM + MW + CDIM (combined K for gates GEMM)
#define G4   2048            // 4*CDIM
#define HOFF 576             // IDIM + MW
#define ODIM 576             // CDIM + MW  (output row)

__device__ __forceinline__ bf16_t to_bf16(float f) {
  unsigned u = __builtin_bit_cast(unsigned, f);
  unsigned r = u + 0x7FFFu + ((u >> 16) & 1u);
  unsigned short h = (unsigned short)(r >> 16);
  return __builtin_bit_cast(bf16_t, h);
}

__device__ __forceinline__ float sigf(float x) { return 1.f / (1.f + __expf(-x)); }

struct U32x8 { uint4 a, b; };

// A fragment (16x32 bf16, M x K): lane L holds row L%16; VGPR0-3 = 8 consecutive
// bf16 at k0 + (L>=16 ? 8 : 0), VGPR4-7 = same + 16.
__device__ __forceinline__ v16bf load_a_frag(const bf16_t* __restrict__ base, int stride, int lane) {
  int m = lane & 15;
  int koff = (lane >> 4) * 8;
  const bf16_t* p = base + (size_t)m * stride + koff;
  U32x8 t;
  t.a = *(const uint4*)(p);
  t.b = *(const uint4*)(p + 16);
  return __builtin_bit_cast(v16bf, t);
}

// B fragment (32x16 bf16, K x N) with B[k][n] = W[n][k], W row-major:
// lane L holds col L%16 = W row n0+L%16; VGPR0-3 = 8 consecutive bf16 at k0 + (L>=16?8:0), VGPR4-7 = same + 16.
__device__ __forceinline__ v16bf load_b_frag(const bf16_t* __restrict__ base, int stride, int lane) {
  int n = lane & 15;
  int koff = (lane >> 4) * 8;
  const bf16_t* p = base + (size_t)n * stride + koff;
  U32x8 t;
  t.a = *(const uint4*)(p);
  t.b = *(const uint4*)(p + 16);
  return __builtin_bit_cast(v16bf, t);
}

// ---------------- init kernels ----------------

__global__ void cvt_embs(const float* __restrict__ e, bf16_t* __restrict__ eb, int n) {
  int i = blockIdx.x * blockDim.x + threadIdx.x;
  if (i < n) eb[i] = to_bf16(e[i]);
}

__global__ void pack_wc(const float* __restrict__ w_ih, const float* __restrict__ w_hh,
                        bf16_t* __restrict__ wc) {
  int i = blockIdx.x * blockDim.x + threadIdx.x;
  if (i >= G4 * KC) return;
  int j = i / KC, k = i - j * KC;
  float v = (k < HOFF) ? w_ih[j * HOFF + k] : w_hh[j * CDIM + (k - HOFF)];
  wc[i] = to_bf16(v);
}

__global__ void pack_wheads(const float* __restrict__ wrk, const float* __restrict__ wwk,
                            const float* __restrict__ we, bf16_t* __restrict__ wh) {
  int i = blockIdx.x * blockDim.x + threadIdx.x;
  if (i >= 192 * CDIM) return;
  int n = i / CDIM, k = i - n * CDIM;
  float v = (n < 64) ? wrk[n * CDIM + k]
          : (n < 128) ? wwk[(n - 64) * CDIM + k]
                      : we[(n - 128) * CDIM + k];
  wh[i] = to_bf16(v);
}

__global__ void pack_wa(const float* __restrict__ wa, bf16_t* __restrict__ o) {
  int i = blockIdx.x * blockDim.x + threadIdx.x;
  if (i < MW * IDIM) o[i] = to_bf16(wa[i]);
}

__global__ void init_state(const float* __restrict__ h0, const float* __restrict__ c0,
                           const float* __restrict__ r0, const float* __restrict__ bih,
                           const float* __restrict__ bhh, float* __restrict__ cst,
                           bf16_t* __restrict__ h0buf, bf16_t* __restrict__ r_bf,
                           float* __restrict__ bc) {
  int idx = blockIdx.x * blockDim.x + threadIdx.x;  // 65536
  int b = idx >> 9, j = idx & 511;
  cst[idx] = c0[j];
  h0buf[(size_t)b * CDIM + j] = to_bf16(h0[j]);
  if (j < MW) r_bf[(size_t)b * MW + j] = to_bf16(r0[j]);
  if (idx < G4) bc[idx] = bih[idx] + bhh[idx];
}

__global__ void init_mem(const float* __restrict__ mem0, float* __restrict__ mem) {
  int i = blockIdx.x * blockDim.x + threadIdx.x;  // 128*65536
  mem[i] = mem0[i & (NMEM * MW - 1)];
}

// a_all = tanh(embs @ W_a^T + b_a)  for all T at once: [32768,512]x[512,64]
__global__ void aall_gemm(const bf16_t* __restrict__ embsb, const bf16_t* __restrict__ Wa,
                          const float* __restrict__ b_a, float* __restrict__ aall) {
  int wave = threadIdx.x >> 5, lane = threadIdx.x & 31;
  int tile = blockIdx.x * 8 + wave;   // 8192 tiles = 2048 m-tiles x 4 n-tiles
  int mt = tile >> 2, nt = tile & 3;
  int m0 = mt * 16, n0 = nt * 16;
  v8f acc = {};
  for (int k = 0; k < IDIM; k += 32) {
    v16bf a = load_a_frag(embsb + (size_t)m0 * IDIM + k, IDIM, lane);
    v16bf b = load_b_frag(Wa + (size_t)n0 * IDIM + k, IDIM, lane);
    acc = __builtin_amdgcn_wmma_f32_16x16x32_bf16(false, a, false, b, (short)0, acc, false, false);
  }
  int n = lane & 15;
  int mh = (lane >> 4) * 8;
#pragma unroll
  for (int i = 0; i < 8; ++i) {
    int mm = m0 + i + mh;
    aall[(size_t)mm * MW + n0 + n] = tanhf(acc[i] + b_a[n0 + n]);
  }
}

// ---------------- per-step kernels ----------------

// Fused gates GEMM + LSTM pointwise.
// Wave `w` owns batch tile m0=w*16; block owns hidden-unit tile j0=blockIdx*16.
// The 4 accumulators are the i/f/g/o gate tiles for the SAME (m,j) patch
// (gate g at Wc rows j0 + g*512), so the LSTM nonlinearity + c/h update fuse
// into the WMMA epilogue. A-matrix K regions: [0,512)=emb_t, [512,576)=r,
// [576,1088)=h_{t-1}; all region boundaries are multiples of 32, so each
// K-tile has a single source.
__global__ void gates_lstm(const bf16_t* __restrict__ emb_t,  // [128,512]
                           const bf16_t* __restrict__ r_bf,   // [128,64]
                           const bf16_t* __restrict__ h_in,   // [128,512] (t-1)
                           const bf16_t* __restrict__ Wc,     // [2048,1088]
                           const float*  __restrict__ bc,     // [2048]
                           float* __restrict__ cst,           // [128,512]
                           bf16_t* __restrict__ h_out,        // [128,512] (t)
                           float* __restrict__ out_t) {       // [128,576]
  int wave = threadIdx.x >> 5, lane = threadIdx.x & 31;
  int m0 = wave * 16;
  int j0 = blockIdx.x * 16;           // 32 blocks cover 512 hidden units
  v8f ai = {}, af = {}, ag = {}, ao = {};
  for (int k = 0; k < KC; k += 32) {
    const bf16_t* abase;
    int astride;
    if (k < IDIM)      { abase = emb_t + (size_t)m0 * IDIM + k;          astride = IDIM; }
    else if (k < HOFF) { abase = r_bf  + (size_t)m0 * MW   + (k - IDIM); astride = MW;   }
    else               { abase = h_in  + (size_t)m0 * CDIM + (k - HOFF); astride = CDIM; }
    v16bf a  = load_a_frag(abase, astride, lane);
    v16bf bi = load_b_frag(Wc + (size_t)(j0 +    0) * KC + k, KC, lane);
    v16bf bf = load_b_frag(Wc + (size_t)(j0 +  512) * KC + k, KC, lane);
    v16bf bg = load_b_frag(Wc + (size_t)(j0 + 1024) * KC + k, KC, lane);
    v16bf bo = load_b_frag(Wc + (size_t)(j0 + 1536) * KC + k, KC, lane);
    ai = __builtin_amdgcn_wmma_f32_16x16x32_bf16(false, a, false, bi, (short)0, ai, false, false);
    af = __builtin_amdgcn_wmma_f32_16x16x32_bf16(false, a, false, bf, (short)0, af, false, false);
    ag = __builtin_amdgcn_wmma_f32_16x16x32_bf16(false, a, false, bg, (short)0, ag, false, false);
    ao = __builtin_amdgcn_wmma_f32_16x16x32_bf16(false, a, false, bo, (short)0, ao, false, false);
  }
  int n = lane & 15;
  int mh = (lane >> 4) * 8;
  int j = j0 + n;
  float bci = bc[j], bcf = bc[512 + j], bcg = bc[1024 + j], bco = bc[1536 + j];
#pragma unroll
  for (int i = 0; i < 8; ++i) {
    int m = m0 + i + mh;
    float ii = sigf (ai[i] + bci);
    float ff = sigf (af[i] + bcf);
    float gg = tanhf(ag[i] + bcg);
    float oo = sigf (ao[i] + bco);
    size_t cidx = (size_t)m * CDIM + j;
    float cn = ff * cst[cidx] + ii * gg;
    cst[cidx] = cn;
    float hn = oo * tanhf(cn);
    h_out[cidx] = to_bf16(hn);
    out_t[(size_t)m * ODIM + j] = hn;
  }
}

// kbuf[b][0:64]=k_r, [64:128]=k_w, [128:192]=sigmoid(e): h_t @ Wh^T, [128,512]x[512,192]
__global__ void heads_gemm(const bf16_t* __restrict__ h_t, const bf16_t* __restrict__ Wh,
                           const float* __restrict__ brk, const float* __restrict__ bwk,
                           const float* __restrict__ be, float* __restrict__ kbuf) {
  int wave = threadIdx.x >> 5, lane = threadIdx.x & 31;
  int m0 = wave * 16;
  int n0 = blockIdx.x * 16;           // 12 blocks cover N=192
  v8f acc = {};
  for (int k = 0; k < CDIM; k += 32) {
    v16bf a = load_a_frag(h_t + (size_t)m0 * CDIM + k, CDIM, lane);
    v16bf b = load_b_frag(Wh + (size_t)n0 * CDIM + k, CDIM, lane);
    acc = __builtin_amdgcn_wmma_f32_16x16x32_bf16(false, a, false, b, (short)0, acc, false, false);
  }
  int n = lane & 15;
  int col = n0 + n;
  int mh = (lane >> 4) * 8;
#pragma unroll
  for (int i = 0; i < 8; ++i) {
    int mm = m0 + i + mh;
    float v = acc[i];
    if (col < 64)       v += brk[col];
    else if (col < 128) v += bwk[col - 64];
    else                v = sigf(v + be[col - 128]);
    kbuf[(size_t)mm * 192 + col] = v;
  }
}

__device__ __forceinline__ float block_max(float v, float* red, int tid) {
  red[tid] = v; __syncthreads();
  for (int s = 128; s > 0; s >>= 1) {
    if (tid < s) red[tid] = fmaxf(red[tid], red[tid + s]);
    __syncthreads();
  }
  float r = red[0]; __syncthreads();
  return r;
}
__device__ __forceinline__ float block_sum(float v, float* red, int tid) {
  red[tid] = v; __syncthreads();
  for (int s = 128; s > 0; s >>= 1) {
    if (tid < s) red[tid] += red[tid + s];
    __syncthreads();
  }
  float r = red[0]; __syncthreads();
  return r;
}

// One block per batch element: content scores, two softmaxes, read vector, erase/add write.
__global__ void mem_step(float* __restrict__ mem, const float* __restrict__ kbuf,
                         const float* __restrict__ aall_t, bf16_t* __restrict__ r_bf,
                         float* __restrict__ out_t) {
  __shared__ float s_r[NMEM];
  __shared__ float s_w[NMEM];
  __shared__ float kr[MW], kw[MW], ev[MW], av[MW];
  __shared__ float red[256];
  int b = blockIdx.x, tid = threadIdx.x;
  if (tid < MW) {
    kr[tid] = kbuf[b * 192 + tid];
    kw[tid] = kbuf[b * 192 + 64 + tid];
    ev[tid] = kbuf[b * 192 + 128 + tid];
    av[tid] = aall_t[b * MW + tid];
  }
  __syncthreads();
  float* memb = mem + (size_t)b * NMEM * MW;

  // phase 1: scores = mem[b] @ k (each thread: 4 rows; a row = 2 cache lines,
  // 8 sequential float4 hits per line -> bytes moved == data size)
  for (int n = tid; n < NMEM; n += 256) {
    const float4* row = (const float4*)(memb + (size_t)n * MW);
    float dr = 0.f, dw = 0.f;
#pragma unroll
    for (int q = 0; q < 16; ++q) {
      float4 v = row[q];
      dr += v.x * kr[4*q] + v.y * kr[4*q+1] + v.z * kr[4*q+2] + v.w * kr[4*q+3];
      dw += v.x * kw[4*q] + v.y * kw[4*q+1] + v.z * kw[4*q+2] + v.w * kw[4*q+3];
    }
    s_r[n] = dr; s_w[n] = dw;
  }
  __syncthreads();

  // softmax over N for both score arrays
  float lm_r = -1e30f, lm_w = -1e30f;
  for (int n = tid; n < NMEM; n += 256) {
    lm_r = fmaxf(lm_r, s_r[n]);
    lm_w = fmaxf(lm_w, s_w[n]);
  }
  float mx_r = block_max(lm_r, red, tid);
  float mx_w = block_max(lm_w, red, tid);
  float ls_r = 0.f, ls_w = 0.f;
  for (int n = tid; n < NMEM; n += 256) {
    float er = __expf(s_r[n] - mx_r); s_r[n] = er; ls_r += er;
    float ew = __expf(s_w[n] - mx_w); s_w[n] = ew; ls_w += ew;
  }
  float inv_r = 1.f / block_sum(ls_r, red, tid);
  float inv_w = 1.f / block_sum(ls_w, red, tid);
  for (int n = tid; n < NMEM; n += 256) {
    s_r[n] *= inv_r;
    s_w[n] *= inv_w;
  }
  __syncthreads();

  // phase 2: r_new = w_r @ mem; mem = mem*(1 - w_w*e) + w_w*a   (coalesced columns)
  int m = tid & 63, seg = tid >> 6;
  float em = ev[m], am = av[m];
  float racc = 0.f;
  for (int n = seg; n < NMEM; n += 4) {
    size_t off = (size_t)n * MW + m;
    float v = memb[off];
    racc += s_r[n] * v;
    float ww = s_w[n];
    memb[off] = v * (1.f - ww * em) + ww * am;
  }
  red[tid] = racc; __syncthreads();
  if (tid < 64) {
    float r = red[tid] + red[tid + 64] + red[tid + 128] + red[tid + 192];
    out_t[(size_t)b * ODIM + CDIM + tid] = r;
    r_bf[(size_t)b * MW + tid] = to_bf16(r);
  }
}

// ---------------- host ----------------

extern "C" void kernel_launch(void* const* d_in, const int* in_sizes, int n_in,
                              void* d_out, int out_size, void* d_ws, size_t ws_size,
                              hipStream_t stream) {
  const float* embs = (const float*)d_in[0];
  const float* w_ih = (const float*)d_in[1];
  const float* w_hh = (const float*)d_in[2];
  const float* b_ih = (const float*)d_in[3];
  const float* b_hh = (const float*)d_in[4];
  const float* h0   = (const float*)d_in[5];
  const float* c0   = (const float*)d_in[6];
  const float* r0   = (const float*)d_in[7];
  const float* mem0 = (const float*)d_in[8];
  const float* W_rk = (const float*)d_in[9];
  const float* b_rk = (const float*)d_in[10];
  const float* W_wk = (const float*)d_in[11];
  const float* b_wk = (const float*)d_in[12];
  const float* W_e  = (const float*)d_in[13];
  const float* b_e  = (const float*)d_in[14];
  const float* W_a  = (const float*)d_in[15];
  const float* b_a  = (const float*)d_in[16];
  float* out = (float*)d_out;

  char* ws = (char*)d_ws;
  size_t off = 0;
  auto take = [&](size_t bytes) -> void* {
    void* p = ws + off;
    off = (off + bytes + 255) & ~(size_t)255;
    return p;
  };
  bf16_t* embs_bf = (bf16_t*)take((size_t)TT * BSZ * IDIM * 2);   // 33.5 MB
  float*  mem     = (float*) take((size_t)BSZ * NMEM * MW * 4);   // 33.5 MB (L2-resident)
  float*  aall    = (float*) take((size_t)TT * BSZ * MW * 4);     // 8.4 MB
  bf16_t* Wc      = (bf16_t*)take((size_t)G4 * KC * 2);           // 4.5 MB
  bf16_t* hbuf0   = (bf16_t*)take((size_t)BSZ * CDIM * 2);        // h ping
  bf16_t* hbuf1   = (bf16_t*)take((size_t)BSZ * CDIM * 2);        // h pong
  bf16_t* r_bf    = (bf16_t*)take((size_t)BSZ * MW * 2);
  float*  cst     = (float*) take((size_t)BSZ * CDIM * 4);
  float*  kbuf    = (float*) take((size_t)BSZ * 192 * 4);
  bf16_t* Wh      = (bf16_t*)take((size_t)192 * CDIM * 2);
  bf16_t* Wa      = (bf16_t*)take((size_t)MW * IDIM * 2);
  float*  bc      = (float*) take((size_t)G4 * 4);

  // one-time packing / precompute
  cvt_embs   <<<(TT * BSZ * IDIM + 255) / 256, 256, 0, stream>>>(embs, embs_bf, TT * BSZ * IDIM);
  pack_wc    <<<(G4 * KC + 255) / 256, 256, 0, stream>>>(w_ih, w_hh, Wc);
  pack_wheads<<<(192 * CDIM + 255) / 256, 256, 0, stream>>>(W_rk, W_wk, W_e, Wh);
  pack_wa    <<<(MW * IDIM + 255) / 256, 256, 0, stream>>>(W_a, Wa);
  init_state <<<(BSZ * CDIM) / 256, 256, 0, stream>>>(h0, c0, r0, b_ih, b_hh, cst, hbuf0, r_bf, bc);
  init_mem   <<<(BSZ * NMEM * MW) / 256, 256, 0, stream>>>(mem0, mem);
  aall_gemm  <<<1024, 256, 0, stream>>>(embs_bf, Wa, b_a, aall);

  for (int t = 0; t < TT; ++t) {
    float* out_t = out + (size_t)t * BSZ * ODIM;
    const bf16_t* emb_t = embs_bf + (size_t)t * BSZ * IDIM;
    bf16_t* h_in  = (t & 1) ? hbuf1 : hbuf0;
    bf16_t* h_out = (t & 1) ? hbuf0 : hbuf1;
    gates_lstm<<<32, 256, 0, stream>>>(emb_t, r_bf, h_in, Wc, bc, cst, h_out, out_t);
    heads_gemm<<<12, 256, 0, stream>>>(h_out, Wh, b_rk, b_wk, b_e, kbuf);
    mem_step  <<<BSZ, 256, 0, stream>>>(mem, kbuf, aall + (size_t)t * BSZ * MW, r_bf, out_t);
  }
}
